// Adapter_48258252538243
// MI455X (gfx1250) — matrix operand order repaired and checked by reading, not compiled, not run on hardware
//
#include <hip/hip_runtime.h>
#include <hip/hip_bf16.h>

// ---------------------------------------------------------------------------
// Problem constants (from reference)
// ---------------------------------------------------------------------------
#define BB 16
#define NN 4096
#define CTX 768
#define AD 1024
#define INNER 320
#define NH 8
#define DH 40
#define DPAD 64          // padded head dim (K multiple of 32 for bf16 WMMA)
#define JJ 64            // audio tokens / A_INNER
#define KS 17
#define CPAD 8
#define LN_EPS 1e-5f

typedef __attribute__((ext_vector_type(16))) __bf16 v16bf;
typedef __attribute__((ext_vector_type(8)))  float  v8f;

// ---------------------------------------------------------------------------
// WMMA helpers (wave32, gfx1250): D = A(16x32 bf16) * B(32x16 bf16) + C(f32)
// Fragment element -> (m,k)/(k,n) mapping: contiguous K, k = 16*(lane/16) + e for both A and B.
// ---------------------------------------------------------------------------
__device__ __forceinline__ v8f wmma_bf(v16bf a, v16bf b, v8f c) {
  return __builtin_amdgcn_wmma_f32_16x16x32_bf16(false, a, false, b,
                                                 (short)0, c, false, false);
}

template <typename F>
__device__ __forceinline__ v16bf make_afrag(int lane, F f) {
  v16bf a;
  int m = lane & 15;
  int hi8 = ((lane >> 4) & 1) << 3;
#pragma unroll
  for (int e = 0; e < 16; ++e) {
    int v = e >> 1, h = e & 1;
    int k = (hi8 << 1) + e;  // contiguous K: 16*half + e
    a[e] = f(m, k);
  }
  return a;
}

// ---------------------------------------------------------------------------
// Kernel 0: f32 -> bf16 weight conversion
// ---------------------------------------------------------------------------
__global__ void cvt_bf16_k(const float* __restrict__ in, __bf16* __restrict__ out, int n) {
  int i = blockIdx.x * 256 + threadIdx.x;
  if (i < n) out[i] = (__bf16)in[i];
}

// ---------------------------------------------------------------------------
// Kernel 1: conv1 (1 -> 64 ch, K=17, same pad) + exact GELU
// ---------------------------------------------------------------------------
__global__ void conv1_gelu_k(const float* __restrict__ audio,
                             const float* __restrict__ w1,
                             const float* __restrict__ b1,
                             float* __restrict__ x1) {
  int b = blockIdx.x >> 6, c = blockIdx.x & 63;
  float w[KS];
#pragma unroll
  for (int t = 0; t < KS; ++t) w[t] = w1[c * KS + t];
  float bias = b1[c];
  const float* in = audio + (size_t)b * AD;
  for (int l = threadIdx.x; l < AD; l += 256) {
    float acc = bias;
#pragma unroll
    for (int t = 0; t < KS; ++t) {
      int p = l + t - CPAD;
      if (p >= 0 && p < AD) acc += w[t] * in[p];
    }
    float g = 0.5f * acc * (1.0f + erff(acc * 0.70710678118654752f));
    x1[((size_t)b * 64 + c) * AD + l] = g;
  }
}

// ---------------------------------------------------------------------------
// Kernel 2: conv2 (64 -> 64 ch)
// ---------------------------------------------------------------------------
__global__ void conv2_k(const float* __restrict__ x1,
                        const float* __restrict__ w2,
                        const float* __restrict__ b2,
                        float* __restrict__ x2) {
  __shared__ float wl[64 * KS];
  int b = blockIdx.x >> 6, co = blockIdx.x & 63;
  for (int i = threadIdx.x; i < 64 * KS; i += 256) wl[i] = w2[(size_t)co * 64 * KS + i];
  __syncthreads();
  for (int l = threadIdx.x; l < AD; l += 256) {
    float acc = b2[co];
    for (int ci = 0; ci < 64; ++ci) {
      const float* row = x1 + ((size_t)b * 64 + ci) * AD;
      const float* ww = wl + ci * KS;
#pragma unroll
      for (int t = 0; t < KS; ++t) {
        int p = l + t - CPAD;
        if (p >= 0 && p < AD) acc += ww[t] * row[p];
      }
    }
    x2[((size_t)b * 64 + co) * AD + l] = acc;
  }
}

// ---------------------------------------------------------------------------
// Kernel 3: LayerNorm statistics over [64,1024] per batch
// ---------------------------------------------------------------------------
__global__ void ln_stats_k(const float* __restrict__ x2, float* __restrict__ stats) {
  __shared__ float s1[256], s2[256];
  int b = blockIdx.x, tid = threadIdx.x;
  const float* base = x2 + (size_t)b * 64 * AD;
  float a = 0.f, aa = 0.f;
  for (int i = tid; i < 64 * AD; i += 256) {
    float v = base[i];
    a += v; aa += v * v;
  }
  s1[tid] = a; s2[tid] = aa;
  __syncthreads();
  for (int s = 128; s > 0; s >>= 1) {
    if (tid < s) { s1[tid] += s1[tid + s]; s2[tid] += s2[tid + s]; }
    __syncthreads();
  }
  if (tid == 0) {
    float mu = s1[0] * (1.0f / (64.0f * AD));
    float var = s2[0] * (1.0f / (64.0f * AD)) - mu * mu;
    stats[b * 2 + 0] = mu;
    stats[b * 2 + 1] = rsqrtf(var + LN_EPS);
  }
}

// ---------------------------------------------------------------------------
// Kernel 4: conv3 over normalized input (LN fused), writes x3
// ---------------------------------------------------------------------------
__global__ void conv3_ln_k(const float* __restrict__ x2,
                           const float* __restrict__ stats,
                           const float* __restrict__ ln_w,
                           const float* __restrict__ ln_b,
                           const float* __restrict__ w3,
                           const float* __restrict__ b3,
                           float* __restrict__ x3) {
  __shared__ float wl[64 * KS];
  int b = blockIdx.x >> 6, co = blockIdx.x & 63;
  for (int i = threadIdx.x; i < 64 * KS; i += 256) wl[i] = w3[(size_t)co * 64 * KS + i];
  __syncthreads();
  float mu = stats[b * 2 + 0], rs = stats[b * 2 + 1];
  for (int l = threadIdx.x; l < AD; l += 256) {
    float acc = b3[co];
    for (int ci = 0; ci < 64; ++ci) {
      const float* row = x2 + ((size_t)b * 64 + ci) * AD;
      const float* gw = ln_w + (size_t)ci * AD;
      const float* gb = ln_b + (size_t)ci * AD;
      const float* ww = wl + ci * KS;
#pragma unroll
      for (int t = 0; t < KS; ++t) {
        int p = l + t - CPAD;
        if (p >= 0 && p < AD) {
          float xv = (row[p] - mu) * rs * gw[p] + gb[p];
          acc += ww[t] * xv;
        }
      }
    }
    x3[((size_t)b * 64 + co) * AD + l] = acc;
  }
}

// ---------------------------------------------------------------------------
// Kernel 5: k/v projection via WMMA.  M=64 (tokens), N=320, K=1024.
// Outputs:
//   kpad: [B][J][NH][DPAD] bf16, zero-padded d in [40,64)  (sim B-fragments)
//   vT:   [B][NH][DH][J]   bf16, j contiguous              (attn*v B-fragments)
// grid = BB, block = 256 (8 waves), dyn LDS = 64*1024 bf16 = 128KB.
// ---------------------------------------------------------------------------
__global__ void kv_proj_k(const float* __restrict__ x3,
                          const __bf16* __restrict__ wkb,
                          const __bf16* __restrict__ wvb,
                          __bf16* __restrict__ kpad,
                          __bf16* __restrict__ vT) {
  extern __shared__ char smem[];
  __bf16* xs = (__bf16*)smem;  // [64][1024]
  int b = blockIdx.x;
  // zero the padded-k slab for this batch (covers d in [40,64))
  for (int i = threadIdx.x; i < JJ * NH * DPAD; i += 256)
    kpad[(size_t)b * JJ * NH * DPAD + i] = (__bf16)0.0f;
  const float* src = x3 + (size_t)b * 64 * AD;
  for (int i = threadIdx.x; i < 64 * AD; i += 256) xs[i] = (__bf16)src[i];
  __syncthreads();
  int lane = threadIdx.x & 31, wave = threadIdx.x >> 5;
  int hi = (lane >> 4) & 1;
  // 160 tiles: [2 (k,v)] x [4 jt] x [20 et]
  for (int t = wave; t < 160; t += 8) {
    int sel = t >= 80;
    int r = t - sel * 80;
    int jt = r / 20, et = r % 20;
    const __bf16* wsel = sel ? wvb : wkb;
    v8f acc = {};
    for (int ksi = 0; ksi < 32; ++ksi) {
      int k0 = ksi * 32;
      v16bf a = make_afrag(lane, [&](int m, int k) {
        return xs[(jt * 16 + m) * AD + k0 + k];
      });
      const v16bf* bp = (const v16bf*)(wsel + (size_t)(et * 16 + (lane & 15)) * AD + k0 + hi * 16);
      acc = wmma_bf(a, *bp, acc);
    }
#pragma unroll
    for (int rr = 0; rr < 8; ++rr) {
      int j = jt * 16 + rr + hi * 8;          // token row
      int n = et * 16 + (lane & 15);          // inner-dim column 0..319
      int h = n / DH, d = n % DH;
      if (sel) {
        vT[(((size_t)b * NH + h) * DH + d) * JJ + j] = (__bf16)acc[rr];
      } else {
        kpad[(((size_t)b * JJ + j) * NH + h) * DPAD + d] = (__bf16)acc[rr];
      }
    }
  }
}

// ---------------------------------------------------------------------------
// Kernel 6: fused q-proj -> sim -> softmax -> attn*v -> out-proj
// grid = BB * (NN/32), block = 256 (8 waves)
// dyn LDS: ctx 32x768 bf16 (48K) + qpad 32x8x64 bf16 (32K) + sim 8x32x64 f32
//          (64K) + attn 8x32x64 bf16 (32K) + out 32x320 bf16 (20K) = 196K
// ---------------------------------------------------------------------------
__global__ void attn_fused_k(const float* __restrict__ context,
                             const __bf16* __restrict__ wqb,
                             const __bf16* __restrict__ kpad,
                             const __bf16* __restrict__ vT,
                             const __bf16* __restrict__ woutb,
                             const float* __restrict__ bout,
                             float* __restrict__ out) {
  extern __shared__ char smem[];
  __bf16* ctx_s = (__bf16*)smem;                     // [32][768]
  __bf16* q_s   = ctx_s + 32 * CTX;                  // [32][NH][DPAD] padded
  float*  sim_s = (float*)(q_s + 32 * NH * DPAD);    // [NH][32][64]
  __bf16* attn_s = (__bf16*)(sim_s + NH * 32 * JJ);  // [NH][32][64]
  __bf16* out_s  = attn_s + NH * 32 * JJ;            // [32][320]

  int b = blockIdx.x >> 7;          // 128 row-tiles per batch
  int row0 = (blockIdx.x & 127) * 32;
  int tid = threadIdx.x, lane = tid & 31, wave = tid >> 5;
  int hi = (lane >> 4) & 1;

  // ---- stage context tile as bf16; zero q padding ----
  const float* cbase = context + ((size_t)b * NN + row0) * CTX;
  for (int i = tid; i < 32 * CTX; i += 256) {
    if ((i & 63) == 0) __builtin_prefetch(cbase + i + 32 * CTX, 0, 1);
    ctx_s[i] = (__bf16)cbase[i];
  }
  for (int i = tid; i < 32 * NH * DPAD; i += 256) q_s[i] = (__bf16)0.0f;
  __syncthreads();

  // ---- q projection: 2 x 20 tiles, K = 768 ----
  for (int t = wave; t < 40; t += 8) {
    int mt = t / 20, et = t % 20;
    v8f acc = {};
    for (int ksi = 0; ksi < 24; ++ksi) {
      int k0 = ksi * 32;
      v16bf a = make_afrag(lane, [&](int m, int k) {
        return ctx_s[(mt * 16 + m) * CTX + k0 + k];
      });
      const v16bf* bp = (const v16bf*)(wqb + (size_t)(et * 16 + (lane & 15)) * CTX + k0 + hi * 16);
      acc = wmma_bf(a, *bp, acc);
    }
#pragma unroll
    for (int rr = 0; rr < 8; ++rr) {
      int m = mt * 16 + rr + hi * 8;
      int n = et * 16 + (lane & 15);
      int h = n / DH, d = n % DH;
      q_s[(m * NH + h) * DPAD + d] = (__bf16)acc[rr];  // padded store
    }
  }
  __syncthreads();

  // ---- sim = (q . k) * scale, per head; padded K = 64, all b128 loads ----
  const float scale = 0.15811388300841897f;  // 40^-0.5
  for (int t = wave; t < 64; t += 8) {
    int h = t >> 3, mt = (t >> 2) & 1, jt = t & 3;
    int j = jt * 16 + (lane & 15);
    const __bf16* krow = kpad + (((size_t)b * JJ + j) * NH + h) * DPAD;
    v8f acc = {};
#pragma unroll
    for (int ksi = 0; ksi < 2; ++ksi) {
      int k0 = ksi * 32;
      v16bf a = make_afrag(lane, [&](int m, int k) {
        return q_s[((mt * 16 + m) * NH + h) * DPAD + k0 + k];
      });
      v16bf bb = *(const v16bf*)(krow + k0 + hi * 16);
      acc = wmma_bf(a, bb, acc);
    }
#pragma unroll
    for (int rr = 0; rr < 8; ++rr) {
      int m = mt * 16 + rr + hi * 8;
      int jn = jt * 16 + (lane & 15);
      sim_s[(h * 32 + m) * JJ + jn] = acc[rr] * scale;
    }
  }
  __syncthreads();

  // ---- softmax over J=64, one (head,row) per thread (256 rows total) ----
  {
    float* row = sim_s + tid * JJ;
    float mx = -1e30f;
#pragma unroll
    for (int j = 0; j < JJ; ++j) mx = fmaxf(mx, row[j]);
    float sm = 0.f;
#pragma unroll
    for (int j = 0; j < JJ; ++j) {
      float e = __expf(row[j] - mx);
      row[j] = e;
      sm += e;
    }
    float inv = 1.0f / sm;
#pragma unroll
    for (int j = 0; j < JJ; ++j) attn_s[tid * JJ + j] = (__bf16)(row[j] * inv);
  }
  __syncthreads();

  // ---- out_h = attn @ v : per head, N=40 via 3 tiles, K=64, vT j-contig ----
  for (int t = wave; t < 48; t += 8) {
    int h = t / 6, rmd = t % 6, mt = rmd / 3, dt = rmd % 3;
    int d = dt * 16 + (lane & 15);
    bool dok = d < DH;
    const __bf16* vrow = vT + (((size_t)b * NH + h) * DH + (dok ? d : 0)) * JJ;
    v8f acc = {};
#pragma unroll
    for (int ksi = 0; ksi < 2; ++ksi) {
      int k0 = ksi * 32;
      v16bf a = make_afrag(lane, [&](int m, int k) {
        return attn_s[(h * 32 + mt * 16 + m) * JJ + k0 + k];
      });
      v16bf bb = *(const v16bf*)(vrow + k0 + hi * 16);   // contiguous in K=j
      if (!dok) bb = (v16bf)(__bf16)0.0f;                // uniform lane mask
      acc = wmma_bf(a, bb, acc);
    }
#pragma unroll
    for (int rr = 0; rr < 8; ++rr) {
      int m = mt * 16 + rr + hi * 8;
      if (dok) out_s[m * INNER + h * DH + d] = (__bf16)acc[rr];
    }
  }
  __syncthreads();

  // ---- final projection: [32 x 320] @ w_out^T -> [32 x 768], + b_out ----
  for (int t = wave; t < 96; t += 8) {
    int mt = t / 48, ct = t % 48;
    const __bf16* wrow = woutb + (size_t)(ct * 16 + (lane & 15)) * INNER;
    __builtin_prefetch(wrow, 0, 1);
    v8f acc = {};
    for (int ksi = 0; ksi < 10; ++ksi) {
      int k0 = ksi * 32;
      v16bf a = make_afrag(lane, [&](int m, int k) {
        return out_s[(mt * 16 + m) * INNER + k0 + k];
      });
      acc = wmma_bf(a, *(const v16bf*)(wrow + k0 + hi * 16), acc);
    }
#pragma unroll
    for (int rr = 0; rr < 8; ++rr) {
      int m = mt * 16 + rr + hi * 8;
      int c = ct * 16 + (lane & 15);
      out[((size_t)b * NN + row0 + m) * CTX + c] = acc[rr] + bout[c];
    }
  }
}

// ---------------------------------------------------------------------------
// Host launcher
// ---------------------------------------------------------------------------
extern "C" void kernel_launch(void* const* d_in, const int* in_sizes, int n_in,
                              void* d_out, int out_size, void* d_ws, size_t ws_size,
                              hipStream_t stream) {
  const float* context = (const float*)d_in[0];
  const float* audio   = (const float*)d_in[1];
  const float* w1 = (const float*)d_in[2];
  const float* b1 = (const float*)d_in[3];
  const float* w2 = (const float*)d_in[4];
  const float* b2 = (const float*)d_in[5];
  const float* ln_w = (const float*)d_in[6];
  const float* ln_b = (const float*)d_in[7];
  const float* w3 = (const float*)d_in[8];
  const float* b3 = (const float*)d_in[9];
  const float* wq = (const float*)d_in[10];
  const float* wk = (const float*)d_in[11];
  const float* wv = (const float*)d_in[12];
  const float* w_out = (const float*)d_in[13];
  const float* b_out = (const float*)d_in[14];
  float* out = (float*)d_out;

  // workspace carve (256B aligned slabs)
  char* wsp = (char*)d_ws;
  size_t off = 0;
  auto carve = [&](size_t bytes) -> char* {
    char* p = wsp + off;
    off += (bytes + 255) & ~(size_t)255;
    return p;
  };
  float* x1    = (float*)carve((size_t)BB * 64 * AD * 4);   // conv1 out (reused for x3)
  float* x2    = (float*)carve((size_t)BB * 64 * AD * 4);   // conv2 out
  float* stats = (float*)carve((size_t)BB * 2 * 4);
  __bf16* wq_bf   = (__bf16*)carve((size_t)INNER * CTX * 2);
  __bf16* wk_bf   = (__bf16*)carve((size_t)INNER * AD * 2);
  __bf16* wv_bf   = (__bf16*)carve((size_t)INNER * AD * 2);
  __bf16* wout_bf = (__bf16*)carve((size_t)CTX * INNER * 2);
  __bf16* kpad = (__bf16*)carve((size_t)BB * JJ * NH * DPAD * 2);
  __bf16* vT   = (__bf16*)carve((size_t)BB * NH * DH * JJ * 2);
  float* x3 = x1;  // x1 dead after conv2; reuse for conv3 output

  // weight conversions (idempotent, deterministic)
  {
    int n;
    n = INNER * CTX;  cvt_bf16_k<<<(n + 255) / 256, 256, 0, stream>>>(wq, wq_bf, n);
    n = INNER * AD;   cvt_bf16_k<<<(n + 255) / 256, 256, 0, stream>>>(wk, wk_bf, n);
    n = INNER * AD;   cvt_bf16_k<<<(n + 255) / 256, 256, 0, stream>>>(wv, wv_bf, n);
    n = CTX * INNER;  cvt_bf16_k<<<(n + 255) / 256, 256, 0, stream>>>(w_out, wout_bf, n);
  }

  // audio encoder
  conv1_gelu_k<<<BB * 64, 256, 0, stream>>>(audio, w1, b1, x1);
  conv2_k<<<BB * 64, 256, 0, stream>>>(x1, w2, b2, x2);
  ln_stats_k<<<BB, 256, 0, stream>>>(x2, stats);
  conv3_ln_k<<<BB * 64, 256, 0, stream>>>(x2, stats, ln_w, ln_b, w3, b3, x3);

  // k/v projection (WMMA), 128KB dynamic LDS
  kv_proj_k<<<BB, 256, (size_t)64 * AD * 2, stream>>>(x3, wk_bf, wv_bf, kpad, vT);

  // fused attention + projections, 196KB dynamic LDS
  size_t smem_attn = (size_t)32 * CTX * 2          // ctx_s
                   + (size_t)32 * NH * DPAD * 2    // q_s (padded)
                   + (size_t)NH * 32 * JJ * 4      // sim_s
                   + (size_t)NH * 32 * JJ * 2      // attn_s
                   + (size_t)32 * INNER * 2;       // out_s
  attn_fused_k<<<BB * (NN / 32), 256, smem_attn, stream>>>(
      context, wq_bf, kpad, vT, wout_bf, b_out, out);
}
